// VisionMambaBlock_74234214744159
// MI455X (gfx1250) — matrix operand order repaired and checked by reading, not compiled, not run on hardware
//
#include <hip/hip_runtime.h>
#include <hip/hip_bf16.h>
#include <cstdint>

#define DIM    96
#define DSTATE 16
#define DCONV  4
#define DIN    192           // EXPAND*DIM
#define DTRANK 6
#define BSZ    2
#define LLEN   4096          // 64*64
#define NTOK   (BSZ*LLEN)    // 8192
#define NXZ    (2*DIN)       // 384
#define NXD    (DTRANK+2*DSTATE) // 38
#define NXDP   64            // padded to 4 WMMA N-tiles (2 pairs)

typedef __attribute__((ext_vector_type(16))) _Float16 v16h;
typedef __attribute__((ext_vector_type(8)))  _Float16 v8h;
typedef __attribute__((ext_vector_type(8)))  float    v8f;

__device__ __forceinline__ float sigmoidf_(float x){ return 1.0f/(1.0f+__expf(-x)); }

// ---------------- weight preparation ----------------
__global__ void k_f32_to_f16(const float* __restrict__ src, _Float16* __restrict__ dst, int n){
  int i = blockIdx.x*blockDim.x + threadIdx.x;
  if (i < n) dst[i] = (_Float16)src[i];
}

__global__ void k_pad_wx(const float* __restrict__ wx, _Float16* __restrict__ dst){
  int i = blockIdx.x*blockDim.x + threadIdx.x; // 64*192
  if (i >= NXDP*DIN) return;
  int n = i / DIN, k = i % DIN;
  dst[i] = (_Float16)(n < NXD ? wx[n*DIN + k] : 0.0f);
}

// Wc[c][d] = sum_j W_proj[c][j] * W_out[j][d]  (96x192), fuses last two GEMMs
__global__ void k_fuse_wout(const float* __restrict__ Wproj, const float* __restrict__ Wout,
                            _Float16* __restrict__ Wc){
  int i = blockIdx.x*blockDim.x + threadIdx.x; // 96*192
  if (i >= DIM*DIN) return;
  int c = i / DIN, d = i % DIN;
  float s = 0.f;
  #pragma unroll 8
  for (int j = 0; j < DIM; ++j) s += Wproj[c*DIM + j] * Wout[j*DIN + d];
  Wc[i] = (_Float16)s;
}

// ---------------- layernorm (wave32 per token; lane owns 3 channels) ----------------
__global__ void k_layernorm(const float* __restrict__ x, const float* __restrict__ g,
                            const float* __restrict__ be, _Float16* __restrict__ xs){
  int wave = (blockIdx.x*blockDim.x + threadIdx.x) >> 5;
  int lane = threadIdx.x & 31;
  if (wave >= NTOK) return;
  int b = wave >> 12, l = wave & (LLEN-1);
  const float* xb = x + (size_t)b*DIM*LLEN + l;       // x[b, c, l] with stride LLEN over c
  float v0 = xb[(size_t)(lane     )*LLEN];
  float v1 = xb[(size_t)(lane + 32)*LLEN];
  float v2 = xb[(size_t)(lane + 64)*LLEN];
  float s  = v0 + v1 + v2;
  float sq = v0*v0 + v1*v1 + v2*v2;
  #pragma unroll
  for (int off = 16; off; off >>= 1){ s += __shfl_xor(s, off, 32); sq += __shfl_xor(sq, off, 32); }
  float mu  = s  * (1.0f/DIM);
  float var = sq * (1.0f/DIM) - mu*mu;
  float r   = rsqrtf(var + 1e-5f);
  _Float16* o = xs + (size_t)wave*DIM;
  o[lane     ] = (_Float16)((v0-mu)*r*g[lane     ] + be[lane     ]);
  o[lane + 32] = (_Float16)((v1-mu)*r*g[lane + 32] + be[lane + 32]);
  o[lane + 64] = (_Float16)((v2-mu)*r*g[lane + 64] + be[lane + 64]);
}

// ---------------- register-blocked WMMA GEMM ----------------
// C[M,N] = A[M,K] * W[N,K]^T.  Block = 8 waves = 256 threads.
// Block computes a 256-row x 32-col panel of C: wave w owns M-tiles {2w, 2w+1},
// all waves share N-tiles {0,1} of the block's 32-column panel.
// The 32xK weight panel is staged into LDS once via async-to-LDS (gfx1250 path),
// then B fragments come from LDS (ds_load_b128), A fragments from global.
// If FUSED: epilogue adds b_proj + residual x and scatters to NCHW output.
template<bool FUSED>
__global__ void __launch_bounds__(256) k_gemm22(
    const _Float16* __restrict__ A, int lda,
    const _Float16* __restrict__ W, int ldw,      // ldw == K (padded)
    float* __restrict__ C, int ldc,               // used when !FUSED
    const float* __restrict__ xin,                // used when FUSED
    const float* __restrict__ bproj,              // used when FUSED
    float* __restrict__ out,                      // used when FUSED
    int MblkPerN, int K){
  __shared__ _Float16 Bs[32*192];                 // up to K=192

  int tid  = threadIdx.x;
  int blk  = blockIdx.x;
  int mblk  = blk % MblkPerN;                     // group of 16 M-tiles
  int npair = blk / MblkPerN;                     // pair of N-tiles
  int nbase = npair*32;

  // ---- stage weight panel W[nbase..nbase+31][0..K) into LDS via ASYNC copies ----
  {
    int cpr = K >> 3;                             // 8-half (16B) chunks per row
    for (int s = tid; s < 32*cpr; s += 256){
      int row = s / cpr, ch = s - row*cpr;
      const _Float16* gp = W + (size_t)(nbase + row)*ldw + ch*8;
      unsigned lds_off = (unsigned)(size_t)&Bs[row*K + ch*8];
      unsigned long long ga = (unsigned long long)(size_t)gp;
      asm volatile("global_load_async_to_lds_b128 %0, %1, off"
                   :: "v"(lds_off), "v"(ga) : "memory");
    }
    asm volatile("s_wait_asynccnt 0x0" ::: "memory");
    __syncthreads();
  }

  int wave = tid >> 5, lane = tid & 31;
  int hi = lane >> 4, lo = lane & 15;

  int mt0  = mblk*16 + wave*2;                    // first of the wave's two M-tiles
  const _Float16* arow0 = A + (size_t)(mt0*16 + lo)*lda;
  const _Float16* arow1 = arow0 + (size_t)16*lda;
  const _Float16* brow0 = &Bs[(     lo)*K];       // N-tile 0, column lo
  const _Float16* brow1 = &Bs[(16 + lo)*K];       // N-tile 1

  v8f acc[2][2] = {};
  for (int kb = 0; kb < K; kb += 32){
    v8h a00 = *(const v8h*)(arow0 + kb +      8*hi);
    v8h a01 = *(const v8h*)(arow0 + kb + 16 + 8*hi);
    v8h a10 = *(const v8h*)(arow1 + kb +      8*hi);
    v8h a11 = *(const v8h*)(arow1 + kb + 16 + 8*hi);
    v16h a0, a1;
    #pragma unroll
    for (int i = 0; i < 8; ++i){ a0[i]=a00[i]; a0[8+i]=a01[i]; a1[i]=a10[i]; a1[8+i]=a11[i]; }
    v8h b00 = *(const v8h*)(brow0 + kb +      8*hi);
    v8h b01 = *(const v8h*)(brow0 + kb + 16 + 8*hi);
    v8h b10 = *(const v8h*)(brow1 + kb +      8*hi);
    v8h b11 = *(const v8h*)(brow1 + kb + 16 + 8*hi);
    v16h b0, b1;
    #pragma unroll
    for (int i = 0; i < 8; ++i){ b0[i]=b00[i]; b0[8+i]=b01[i]; b1[i]=b10[i]; b1[8+i]=b11[i]; }
    acc[0][0] = __builtin_amdgcn_wmma_f32_16x16x32_f16(false, a0, false, b0, (short)0, acc[0][0], false, false);
    acc[0][1] = __builtin_amdgcn_wmma_f32_16x16x32_f16(false, a0, false, b1, (short)0, acc[0][1], false, false);
    acc[1][0] = __builtin_amdgcn_wmma_f32_16x16x32_f16(false, a1, false, b0, (short)0, acc[1][0], false, false);
    acc[1][1] = __builtin_amdgcn_wmma_f32_16x16x32_f16(false, a1, false, b1, (short)0, acc[1][1], false, false);
  }

  if (!FUSED){
    #pragma unroll
    for (int it = 0; it < 2; ++it){
      #pragma unroll
      for (int jt = 0; jt < 2; ++jt){
        float* cbase = C + (size_t)((mt0+it)*16 + 8*hi)*ldc + nbase + jt*16 + lo;
        #pragma unroll
        for (int r = 0; r < 8; ++r) cbase[(size_t)r*ldc] = acc[it][jt][r];
      }
    }
  } else {
    #pragma unroll
    for (int jt = 0; jt < 2; ++jt){
      int c = nbase + jt*16 + lo;
      float bp = bproj[c];
      #pragma unroll
      for (int it = 0; it < 2; ++it){
        #pragma unroll
        for (int r = 0; r < 8; ++r){
          int m = (mt0+it)*16 + r + 8*hi;          // token index
          int b = m >> 12, l = m & (LLEN-1);
          size_t gi = (((size_t)b*DIM + c) << 12) + l;
          out[gi] = xin[gi] + bp + acc[it][jt][r];
        }
      }
    }
  }
}

// ---------------- depthwise causal conv(4) + SiLU ----------------
__global__ void k_conv_silu(const float* __restrict__ xz, const float* __restrict__ cw,
                            const float* __restrict__ cb,
                            float* __restrict__ u32, _Float16* __restrict__ u16){
  int i = blockIdx.x*blockDim.x + threadIdx.x;
  if (i >= NTOK*DIN) return;
  int d = i % DIN, t = i / DIN;
  int l = t & (LLEN-1), tb = t - l;
  float s = cb[d];
  #pragma unroll
  for (int j = 0; j < DCONV; ++j){
    int ll = l - (DCONV-1) + j;
    if (ll >= 0) s += cw[d*DCONV + j] * xz[(size_t)(tb + ll)*NXZ + d];
  }
  float uu = s * sigmoidf_(s);
  u32[i] = uu;
  u16[i] = (_Float16)uu;
}

// ---------------- delta = softplus(dt_raw @ W_dt^T + b_dt), K=6 scalar ----------------
__global__ void k_delta(const float* __restrict__ xdbl, const float* __restrict__ Wdt,
                        const float* __restrict__ bdt, float* __restrict__ delta){
  int i = blockIdx.x*blockDim.x + threadIdx.x;
  if (i >= NTOK*DIN) return;
  int d = i % DIN, t = i / DIN;
  const float* xr = xdbl + (size_t)t*NXDP;
  float s = bdt[d];
  #pragma unroll
  for (int r = 0; r < DTRANK; ++r) s += xr[r] * Wdt[d*DTRANK + r];
  delta[i] = (s > 20.f) ? s : log1pf(__expf(s));
}

// ---------------- selective scan: thread per (batch, channel), B/C via LDS ----------------
__global__ void __launch_bounds__(DIN) k_scan(const float* __restrict__ xdbl,
                                              const float* __restrict__ delta,
                                              const float* __restrict__ u32,
                                              const float* __restrict__ xz,
                                              const float* __restrict__ Alog,
                                              const float* __restrict__ Dp,
                                              _Float16* __restrict__ y16){
  __shared__ float BC[2*DSTATE];
  int d = threadIdx.x, b = blockIdx.x;
  float Ad[DSTATE], h[DSTATE];
  #pragma unroll
  for (int n = 0; n < DSTATE; ++n){ Ad[n] = -__expf(Alog[d*DSTATE + n]); h[n] = 0.f; }
  float Dd = Dp[d];
  for (int l = 0; l < LLEN; ++l){
    int t = b*LLEN + l;
    __syncthreads();
    if (d < 2*DSTATE) BC[d] = xdbl[(size_t)t*NXDP + DTRANK + d];
    __syncthreads();
    float dt = delta[(size_t)t*DIN + d];
    float uu = u32  [(size_t)t*DIN + d];
    float y = 0.f;
    #pragma unroll
    for (int n = 0; n < DSTATE; ++n){
      float dA = __expf(dt*Ad[n]);
      h[n] = dA*h[n] + dt*BC[n]*uu;
      y  += h[n]*BC[DSTATE + n];
    }
    y += uu*Dd;
    float zz = xz[(size_t)t*NXZ + DIN + d];
    y *= zz * sigmoidf_(zz);                      // * silu(z)
    y16[(size_t)t*DIN + d] = (_Float16)y;
  }
}

// ---------------- launcher ----------------
extern "C" void kernel_launch(void* const* d_in, const int* in_sizes, int n_in,
                              void* d_out, int out_size, void* d_ws, size_t ws_size,
                              hipStream_t stream){
  const float* x      = (const float*)d_in[0];
  const float* gamma  = (const float*)d_in[1];
  const float* beta   = (const float*)d_in[2];
  const float* W_in   = (const float*)d_in[3];
  const float* conv_w = (const float*)d_in[4];
  const float* conv_b = (const float*)d_in[5];
  const float* W_x    = (const float*)d_in[6];
  const float* W_dt   = (const float*)d_in[7];
  const float* b_dt   = (const float*)d_in[8];
  const float* A_log  = (const float*)d_in[9];
  const float* Dp     = (const float*)d_in[10];
  const float* W_out  = (const float*)d_in[11];
  const float* W_proj = (const float*)d_in[12];
  const float* b_proj = (const float*)d_in[13];
  float* out = (float*)d_out;

  char* ws = (char*)d_ws;
  size_t off = 0;
  auto alloc = [&](size_t bytes)->char*{
    char* p = ws + off; off = (off + bytes + 255) & ~(size_t)255; return p;
  };
  _Float16* xs16  = (_Float16*)alloc((size_t)NTOK*DIM *2);
  float*    xz    = (float*)   alloc((size_t)NTOK*NXZ *4);
  float*    u32   = (float*)   alloc((size_t)NTOK*DIN *4);
  _Float16* u16   = (_Float16*)alloc((size_t)NTOK*DIN *2);
  float*    xdbl  = (float*)   alloc((size_t)NTOK*NXDP*4);
  float*    dlt   = (float*)   alloc((size_t)NTOK*DIN *4);
  _Float16* y16   = (_Float16*)alloc((size_t)NTOK*DIN *2);
  _Float16* Win16 = (_Float16*)alloc((size_t)NXZ *DIM *2);
  _Float16* Wx16  = (_Float16*)alloc((size_t)NXDP*DIN *2);
  _Float16* Wc16  = (_Float16*)alloc((size_t)DIM *DIN *2);

  // weight prep
  k_f32_to_f16<<<(NXZ*DIM + 255)/256, 256, 0, stream>>>(W_in, Win16, NXZ*DIM);
  k_pad_wx    <<<(NXDP*DIN + 255)/256, 256, 0, stream>>>(W_x, Wx16);
  k_fuse_wout <<<(DIM*DIN + 255)/256, 256, 0, stream>>>(W_proj, W_out, Wc16);

  // layernorm -> xs16 (8192 x 96 f16), 8 waves/block
  k_layernorm<<<NTOK/8, 256, 0, stream>>>(x, gamma, beta, xs16);

  // xz = xs @ W_in^T : M=8192 N=384 K=96 ; 256x32 panel per block
  { int MblkPerN = (NTOK/16)/16, nPairs = (NXZ/16)/2;      // 32, 12
    k_gemm22<false><<<MblkPerN*nPairs, 256, 0, stream>>>(
        xs16, DIM, Win16, DIM, xz, NXZ, nullptr, nullptr, nullptr, MblkPerN, DIM); }

  // u = silu(conv(xi) + b)
  k_conv_silu<<<(NTOK*DIN + 255)/256, 256, 0, stream>>>(xz, conv_w, conv_b, u32, u16);

  // x_dbl = u @ W_x^T : M=8192 N=64(pad) K=192
  { int MblkPerN = (NTOK/16)/16, nPairs = (NXDP/16)/2;     // 32, 2
    k_gemm22<false><<<MblkPerN*nPairs, 256, 0, stream>>>(
        u16, DIN, Wx16, DIN, xdbl, NXDP, nullptr, nullptr, nullptr, MblkPerN, DIN); }

  // delta = softplus(dt_raw @ W_dt^T + b_dt)
  k_delta<<<(NTOK*DIN + 255)/256, 256, 0, stream>>>(xdbl, W_dt, b_dt, dlt);

  // selective scan + gating -> y16
  k_scan<<<BSZ, DIN, 0, stream>>>(xdbl, dlt, u32, xz, A_log, Dp, y16);

  // out = x + y @ (W_proj W_out)^T + b_proj, scattered to NCHW
  { int MblkPerN = (NTOK/16)/16, nPairs = (DIM/16)/2;      // 32, 3
    k_gemm22<true><<<MblkPerN*nPairs, 256, 0, stream>>>(
        y16, DIN, Wc16, DIN, nullptr, 0, x, b_proj, out, MblkPerN, DIN); }
}
